// ProteinFeatures_51548197486887
// MI455X (gfx1250) — matrix operand order repaired and checked by reading, not compiled, not run on hardware
//
#include <hip/hip_runtime.h>
#include <math.h>

#define B_    4
#define N_    2048
#define TOPK  30
#define EDGE_F 128
#define NODE_F 128
#define FSTR  264   // padded half stride for 256-wide feature rows
#define OSTR  132   // padded float stride for 128-wide output rows
#define WAVES 8     // waves per edge block
#define ITER  2     // edge-tile iterations per wave

typedef int v4i __attribute__((ext_vector_type(4)));
typedef __attribute__((address_space(1))) v4i* gas_v4i_p;
typedef __attribute__((address_space(3))) v4i* las_v4i_p;

#if __has_builtin(__builtin_amdgcn_global_load_async_to_lds_b128)
#define HAVE_ASYNC_LDS 1
#if __has_builtin(__builtin_amdgcn_s_wait_asynccnt)
#define WAIT_ASYNC() __builtin_amdgcn_s_wait_asynccnt(0)
#else
#define WAIT_ASYNC() asm volatile("s_wait_asynccnt 0" ::: "memory")
#endif
#endif

typedef __attribute__((ext_vector_type(16))) _Float16 v16h;
typedef __attribute__((ext_vector_type(8)))  float    v8f;

__device__ __forceinline__ void xnorm3(const float* a, float* o) {
    float n = sqrtf(a[0]*a[0] + a[1]*a[1] + a[2]*a[2]);
    float s = 1.0f / fmaxf(n, 1e-12f);
    o[0] = a[0]*s; o[1] = a[1]*s; o[2] = a[2]*s;
}
__device__ __forceinline__ void xcross3(const float* a, const float* b, float* o) {
    o[0] = a[1]*b[2] - a[2]*b[1];
    o[1] = a[2]*b[0] - a[0]*b[2];
    o[2] = a[0]*b[1] - a[1]*b[0];
}
__device__ __forceinline__ float xdot3(const float* a, const float* b) {
    return a[0]*b[0] + a[1]*b[1] + a[2]*b[2];
}

// ---------------------------------------------------------------------------
// Kernel 0: pre-convert W_edge (128x256 f32) -> f16 in padded LDS-image layout
// ---------------------------------------------------------------------------
__global__ void wconv_kernel(const float* __restrict__ We,
                             _Float16* __restrict__ Wh) {
    int idx = blockIdx.x * blockDim.x + threadIdx.x;
    if (idx >= EDGE_F * FSTR) return;
    int n = idx / FSTR, k = idx - n * FSTR;
    float v = (k < 256) ? We[n * 256 + k] : 0.0f;
    Wh[idx] = (_Float16)v;
}

// ---------------------------------------------------------------------------
// Kernel 1: per-node prep — Cb coordinates (to ws) + dihedral node features
//           -> V = LN(feat @ W_node^T)
// ---------------------------------------------------------------------------
__global__ void prep_kernel(const float* __restrict__ X,
                            const float* __restrict__ dmask,
                            const float* __restrict__ Wn,
                            const float* __restrict__ gn,
                            const float* __restrict__ bn,
                            float* __restrict__ Cb_ws,
                            float* __restrict__ Vout) {
    int g = blockIdx.x * blockDim.x + threadIdx.x;
    if (g >= B_ * N_) return;
    int b = g / N_;
    int i = g - b * N_;
    const float* xb = X + (size_t)b * N_ * 12;

    // --- Cb ---
    float bv[3], cv[3], av[3];
    #pragma unroll
    for (int c = 0; c < 3; ++c) {
        float Nn = xb[(size_t)i*12 + 0*3 + c];
        float Ca = xb[(size_t)i*12 + 1*3 + c];
        float Cc = xb[(size_t)i*12 + 2*3 + c];
        bv[c] = Ca - Nn;
        cv[c] = Cc - Ca;
    }
    xcross3(bv, cv, av);
    #pragma unroll
    for (int c = 0; c < 3; ++c) {
        float Ca = xb[(size_t)i*12 + 1*3 + c];
        Cb_ws[(size_t)g*3 + c] = -0.58273431f*av[c] + 0.56802827f*bv[c]
                                 - 0.54067466f*cv[c] + Ca;
    }

    // --- dihedrals ---
    float feat[6];
    #pragma unroll
    for (int a = 0; a < 3; ++a) {
        int p = 3*i + a;
        float ang = 0.0f;
        if (p >= 1 && p <= 3*N_ - 3) {
            int t = p - 1;
            float P[4][3];
            #pragma unroll
            for (int q = 0; q < 4; ++q) {
                int fq = t + q;
                int nd = fq / 3, at = fq - nd*3;
                #pragma unroll
                for (int c = 0; c < 3; ++c)
                    P[q][c] = xb[(size_t)nd*12 + at*3 + c];
            }
            float d0[3], d1[3], d2[3], u2[3], u1[3], u0[3];
            #pragma unroll
            for (int c = 0; c < 3; ++c) {
                d0[c] = P[1][c] - P[0][c];
                d1[c] = P[2][c] - P[1][c];
                d2[c] = P[3][c] - P[2][c];
            }
            xnorm3(d0, u2); xnorm3(d1, u1); xnorm3(d2, u0);
            float c21[3], c10[3], n2[3], n1[3];
            xcross3(u2, u1, c21); xcross3(u1, u0, c10);
            xnorm3(c21, n2); xnorm3(c10, n1);
            float cd = xdot3(n2, n1);
            cd = fminf(fmaxf(cd, -1.0f + 1e-7f), 1.0f - 1e-7f);
            float sg = xdot3(u2, n1);
            float sgn = (sg > 0.0f) ? 1.0f : ((sg < 0.0f) ? -1.0f : 0.0f);
            ang = sgn * acosf(cd);
        }
        float dm = dmask[(size_t)g*3 + a];
        feat[a]     = cosf(ang) * dm;
        feat[3 + a] = sinf(ang) * dm;
    }

    // --- V = LN(feat @ Wn^T) ---
    float sum = 0.0f, ss = 0.0f;
    for (int o = 0; o < NODE_F; ++o) {
        float y = 0.0f;
        #pragma unroll
        for (int c = 0; c < 6; ++c) y += feat[c] * Wn[o*6 + c];
        sum += y; ss += y*y;
    }
    float mean = sum * (1.0f / NODE_F);
    float var  = ss * (1.0f / NODE_F) - mean*mean;
    float rinv = rsqrtf(var + 1e-5f);
    float* dst = Vout + (size_t)g * NODE_F;
    for (int o = 0; o < NODE_F; ++o) {
        float y = 0.0f;
        #pragma unroll
        for (int c = 0; c < 6; ++c) y += feat[c] * Wn[o*6 + c];
        dst[o] = gn[o] * (y - mean) * rinv + bn[o];
    }
}

// ---------------------------------------------------------------------------
// Kernel 2: masked distance + top-30 per (b,i); wave-level butterfly argmin.
// Order: ascending D_adj, ties -> lower index (matches jax.lax.top_k on -D).
// ---------------------------------------------------------------------------
__global__ void topk_kernel(const float* __restrict__ X,
                            const float* __restrict__ mask,
                            int*   __restrict__ Eidx_ws,
                            float* __restrict__ Dnb_ws,
                            float* __restrict__ EidxOut) {
    __shared__ float sD[N_];
    __shared__ float sRv[8];
    __shared__ int   sRi[8];
    __shared__ float sDmax;
    int row = blockIdx.x;
    int b = row / N_;
    int i = row - b * N_;
    int tid = threadIdx.x;
    int lane = tid & 31;
    int wv = tid >> 5;
    const float* xb = X + (size_t)b * N_ * 12;
    float cax = xb[(size_t)i*12 + 3], cay = xb[(size_t)i*12 + 4], caz = xb[(size_t)i*12 + 5];
    float mi = mask[row];

    float lmax = -1.0f;
    for (int j = tid; j < N_; j += 256) {
        float dx = cax - xb[(size_t)j*12 + 3];
        float dy = cay - xb[(size_t)j*12 + 4];
        float dz = caz - xb[(size_t)j*12 + 5];
        float m2 = mi * mask[(size_t)b*N_ + j];
        float D = m2 * sqrtf(dx*dx + dy*dy + dz*dz + 1e-6f);
        sD[j] = D;
        lmax = fmaxf(lmax, D);
    }
    #pragma unroll
    for (int s = 16; s > 0; s >>= 1) lmax = fmaxf(lmax, __shfl_xor(lmax, s, 32));
    if (lane == 0) sRv[wv] = lmax;
    __syncthreads();
    if (tid == 0) {
        float m = sRv[0];
        #pragma unroll
        for (int w = 1; w < 8; ++w) m = fmaxf(m, sRv[w]);
        sDmax = m;
    }
    __syncthreads();
    float Dmax = sDmax;
    for (int j = tid; j < N_; j += 256) {
        float m2 = mi * mask[(size_t)b*N_ + j];
        sD[j] += (1.0f - m2) * Dmax;
    }
    __syncthreads();

    for (int t = 0; t < TOPK; ++t) {
        float v = 3.4e38f; int idx = 0x7fffffff;
        for (int j = tid; j < N_; j += 256) {
            float d = sD[j];
            if (d < v) { v = d; idx = j; }   // ascending j: smallest idx on tie
        }
        #pragma unroll
        for (int s = 16; s > 0; s >>= 1) {
            float v2 = __shfl_xor(v, s, 32);
            int   i2 = __shfl_xor(idx, s, 32);
            if (v2 < v || (v2 == v && i2 < idx)) { v = v2; idx = i2; }
        }
        if (lane == 0) { sRv[wv] = v; sRi[wv] = idx; }
        __syncthreads();
        if (tid == 0) {
            float bvv = sRv[0]; int bii = sRi[0];
            #pragma unroll
            for (int w = 1; w < 8; ++w) {
                if (sRv[w] < bvv || (sRv[w] == bvv && sRi[w] < bii)) {
                    bvv = sRv[w]; bii = sRi[w];
                }
            }
            Eidx_ws[(size_t)row*TOPK + t] = bii;
            Dnb_ws [(size_t)row*TOPK + t] = bvv;
            EidxOut[(size_t)row*TOPK + t] = (float)bii;
            sD[bii] = 3.4e38f;
        }
        __syncthreads();
    }
}

// ---------------------------------------------------------------------------
// Kernel 3: fused edge features (PE + 15x16 RBF) -> WMMA GEMM 256->128 -> LN
// 256 threads = 8 waves; wave owns a 16-edge tile; ITER tiles per wave.
// W staged once per block via async global->LDS DMA (overlaps feature build).
// ---------------------------------------------------------------------------
__global__ void edge_kernel(const float* __restrict__ X,
                            const int*   __restrict__ ridx,
                            const int*   __restrict__ chain,
                            const _Float16* __restrict__ Wh,   // preconverted, FSTR layout
                            const float* __restrict__ Wpe,
                            const float* __restrict__ bpe,
                            const float* __restrict__ ge,
                            const float* __restrict__ be,
                            const float* __restrict__ Cb_ws,
                            const int*   __restrict__ Eidx_ws,
                            const float* __restrict__ Dnb_ws,
                            float* __restrict__ Eout) {
    extern __shared__ char smem[];
    _Float16* sW = (_Float16*)smem;                              // [128][FSTR] f16
    int tid  = threadIdx.x;
    int wave = tid >> 5;
    int lane = tid & 31;

    // --- stage W into LDS (async DMA, ASYNCcnt-tracked) ---
#ifdef HAVE_ASYNC_LDS
    {
        gas_v4i_p gsrc = (gas_v4i_p)Wh;
        las_v4i_p ldst = (las_v4i_p)smem;
        for (int idx = tid; idx < (EDGE_F * FSTR * 2) / 16; idx += 256)
            __builtin_amdgcn_global_load_async_to_lds_b128(gsrc + idx, ldst + idx, 0, 0);
    }
#else
    {
        const unsigned* gsrc = (const unsigned*)Wh;
        unsigned* ldst = (unsigned*)smem;
        for (int idx = tid; idx < EDGE_F * FSTR / 2; idx += 256)
            ldst[idx] = gsrc[idx];
    }
#endif

    char* scr = smem + (size_t)EDGE_F * FSTR * 2;
    _Float16* sFh  = (_Float16*)(scr + (size_t)wave * (16 * FSTR * 2)); // 16x256 f16
    float*    sOut = (float*)   (scr + (size_t)wave * (16 * FSTR * 2)); // aliased 16x128 f32

    int m  = lane & 15;
    int h2 = lane >> 4;
    const int pa[15] = {1,0,2,3,4,1,1,1,1,0,0,0,4,4,2};
    const int pb[15] = {1,0,2,3,4,0,2,3,4,2,3,4,2,3,3};

    for (int it = 0; it < ITER; ++it) {
        int tile = blockIdx.x * (WAVES * ITER) + it * WAVES + wave;
        int e  = tile * 16 + m;
        int b  = e / (N_ * TOPK);
        int rem = e - b * (N_ * TOPK);
        int i  = rem / TOPK;
        int j  = Eidx_ws[e];
        const float* xb = X + (size_t)b * N_ * 12;

        // atoms: 0=N 1=Ca 2=C 3=O 4=Cb
        float Pi[5][3], Pj[5][3];
        #pragma unroll
        for (int a = 0; a < 4; ++a)
            #pragma unroll
            for (int c = 0; c < 3; ++c) {
                Pi[a][c] = xb[(size_t)i*12 + a*3 + c];
                Pj[a][c] = xb[(size_t)j*12 + a*3 + c];
            }
        #pragma unroll
        for (int c = 0; c < 3; ++c) {
            Pi[4][c] = Cb_ws[((size_t)b*N_ + i)*3 + c];
            Pj[4][c] = Cb_ws[((size_t)b*N_ + j)*3 + c];
        }

        // positional embedding -> k = 0..15
        {
            int off = ridx[(size_t)b*N_ + i] - ridx[(size_t)b*N_ + j];
            int cm  = (chain[(size_t)b*N_ + i] == chain[(size_t)b*N_ + j]) ? 1 : 0;
            int dpe = off + 32;
            dpe = dpe < 0 ? 0 : (dpe > 64 ? 64 : dpe);
            dpe = cm ? dpe : 65;
            #pragma unroll
            for (int c = 0; c < 8; ++c) {
                int ci = h2*8 + c;
                sFh[m*FSTR + ci] = (_Float16)(Wpe[ci*66 + dpe] + bpe[ci]);
            }
        }

        // 15 RBF groups -> k = 16..255 (group 0 from D_neighbors)
        #pragma unroll
        for (int g = 0; g < 15; ++g) {
            float d;
            if (g == 0) {
                d = Dnb_ws[e];
            } else {
                float dx = Pi[pa[g]][0] - Pj[pb[g]][0];
                float dy = Pi[pa[g]][1] - Pj[pb[g]][1];
                float dz = Pi[pa[g]][2] - Pj[pb[g]][2];
                d = sqrtf(dx*dx + dy*dy + dz*dz + 1e-6f);
            }
            #pragma unroll
            for (int c = 0; c < 8; ++c) {
                int ci = h2*8 + c;
                float mu = 2.0f + (20.0f / 15.0f) * (float)ci;  // linspace(2,22,16)
                float t  = (d - mu) * 0.8f;                     // / sigma(=1.25)
                sFh[m*FSTR + 16 + g*16 + ci] = (_Float16)__expf(-t*t);
            }
        }

        if (it == 0) {
#ifdef HAVE_ASYNC_LDS
            WAIT_ASYNC();             // W DMA complete (overlapped with features)
#endif
            __syncthreads();          // sW visible to all waves
        } else {
            asm volatile("" ::: "memory");   // order feat stores vs fragment loads
        }

        // WMMA: D[16x128] = A[16x256] * B[256x128], f16 in / f32 accumulate
        v8f acc[8];
        v8f zero = {0.f,0.f,0.f,0.f,0.f,0.f,0.f,0.f};
        #pragma unroll
        for (int nt = 0; nt < 8; ++nt) acc[nt] = zero;

        int offA = h2 * 16;
        #pragma unroll
        for (int kk = 0; kk < 8; ++kk) {
            union { v16h vec; unsigned u[8]; } A;
            const _Float16* fb = sFh + m*FSTR + kk*32;
            #pragma unroll
            for (int r = 0; r < 8; ++r) {
                int k = offA + 2*r;
                A.u[r] = *(const unsigned*)(fb + k);
            }
            #pragma unroll
            for (int nt = 0; nt < 8; ++nt) {
                union { v16h vec; unsigned u[8]; } Bm;
                const _Float16* wb = sW + ((lane & 15) + 16*nt) * FSTR + kk*32 + h2*16;
                #pragma unroll
                for (int r = 0; r < 8; ++r) Bm.u[r] = *(const unsigned*)(wb + 2*r);
                acc[nt] = __builtin_amdgcn_wmma_f32_16x16x32_f16(
                    false, A.vec, false, Bm.vec, (short)0, acc[nt], false, false);
            }
        }

        // scatter D into per-wave LDS scratch (aliases dead feature buffer)
        #pragma unroll
        for (int nt = 0; nt < 8; ++nt)
            #pragma unroll
            for (int r = 0; r < 8; ++r)
                sOut[(h2*8 + r) * OSTR + (lane & 15) + 16*nt] = acc[nt][r];
        asm volatile("" ::: "memory");

        // layer norm over 128 outputs; two lanes per edge (64 cols each)
        float sum = 0.0f, ss = 0.0f;
        const float* orow = sOut + m * OSTR + h2 * 64;
        #pragma unroll 4
        for (int c = 0; c < 64; ++c) {
            float y = orow[c];
            sum += y; ss += y*y;
        }
        sum += __shfl_xor(sum, 16, 32);
        ss  += __shfl_xor(ss, 16, 32);
        float mean = sum * (1.0f / EDGE_F);
        float var  = ss * (1.0f / EDGE_F) - mean*mean;
        float rinv = rsqrtf(var + 1e-5f);
        float* dst = Eout + (size_t)e * EDGE_F + h2 * 64;
        #pragma unroll 4
        for (int c = 0; c < 64; ++c) {
            int n = h2*64 + c;
            float y = orow[c];
            dst[c] = ge[n] * (y - mean) * rinv + be[n];
        }
        asm volatile("" ::: "memory");   // LN reads done before next-iter stores
    }
}

// ---------------------------------------------------------------------------
extern "C" void kernel_launch(void* const* d_in, const int* in_sizes, int n_in,
                              void* d_out, int out_size, void* d_ws, size_t ws_size,
                              hipStream_t stream) {
    const float* X     = (const float*)d_in[0];
    const float* mask  = (const float*)d_in[2];
    const int*   ridx  = (const int*)  d_in[3];
    const float* dmask = (const float*)d_in[4];
    const int*   chain = (const int*)  d_in[5];
    const float* Wn    = (const float*)d_in[6];
    const float* We    = (const float*)d_in[7];
    const float* Wpe   = (const float*)d_in[8];
    const float* bpe   = (const float*)d_in[9];
    const float* gn    = (const float*)d_in[10];
    const float* bn    = (const float*)d_in[11];
    const float* ge    = (const float*)d_in[12];
    const float* be    = (const float*)d_in[13];

    float* out     = (float*)d_out;
    float* Vout    = out;                                       // B*N*128
    float* Eout    = Vout + (size_t)B_*N_*NODE_F;               // B*N*30*128
    float* EidxOut = Eout + (size_t)B_*N_*TOPK*EDGE_F;          // B*N*30 (as float)

    float*    Cb_ws   = (float*)d_ws;                            // B*N*3 floats
    int*      Eidx_ws = (int*)(Cb_ws + (size_t)B_*N_*3);         // B*N*30 ints
    float*    Dnb_ws  = (float*)(Eidx_ws + (size_t)B_*N_*TOPK);  // B*N*30 floats
    _Float16* Wh_ws   = (_Float16*)(Dnb_ws + (size_t)B_*N_*TOPK);// 128*FSTR halves

    wconv_kernel<<<(EDGE_F*FSTR + 255)/256, 256, 0, stream>>>(We, Wh_ws);
    prep_kernel<<<(B_*N_ + 255)/256, 256, 0, stream>>>(X, dmask, Wn, gn, bn, Cb_ws, Vout);
    topk_kernel<<<B_*N_, 256, 0, stream>>>(X, mask, Eidx_ws, Dnb_ws, EidxOut);

    size_t smem = (size_t)EDGE_F * FSTR * 2 + (size_t)WAVES * (16 * FSTR * 2); // 135168 B
    int nblocks = (B_*N_*TOPK) / (16 * WAVES * ITER);                          // 960
    edge_kernel<<<nblocks, 32 * WAVES, smem, stream>>>(X, ridx, chain, Wh_ws,
                                                       Wpe, bpe, ge, be,
                                                       Cb_ws, Eidx_ws, Dnb_ws, Eout);
}
